// SwinTransformerBlock1D_86242943303938
// MI455X (gfx1250) — matrix operand order repaired and checked by reading, not compiled, not run on hardware
//
#include <hip/hip_runtime.h>
#include <stdint.h>

// ---------------- types ----------------
typedef _Float16 v8h  __attribute__((ext_vector_type(8)));
typedef _Float16 v16h __attribute__((ext_vector_type(16)));
typedef float    v8f  __attribute__((ext_vector_type(8)));

#define WMMA_F16(a,b,c) __builtin_amdgcn_wmma_f32_16x16x32_f16(false,(a),false,(b),(short)0,(c),false,false)

// Async global->LDS 16B copy (per lane). LDS address = low 32 bits of generic ptr.
__device__ __forceinline__ void async_b128(void* lds_dst, const void* gsrc) {
    uint32_t d = (uint32_t)(uintptr_t)lds_dst;
    uint64_t s = (uint64_t)(uintptr_t)gsrc;
    asm volatile("global_load_async_to_lds_b128 %0, %1, off" :: "v"(d), "v"(s) : "memory");
}
__device__ __forceinline__ void wait_async0() {
    asm volatile("s_wait_asynccnt 0" ::: "memory");
}

// A-fragment (16x32 f16): lane L holds row L&15; VGPR 0-3 K=(L>=16?8:0)+0..7, VGPR 4-7 K=+16.
__device__ __forceinline__ v16h ld_afrag(const _Float16* p, int lane) {
    int kh = ((lane >> 4) << 3);                       // 0 or 8
    v8h lo = *(const v8h*)(p + kh);
    v8h hi = *(const v8h*)(p + 16 + kh);
    return __builtin_shufflevector(lo, hi, 0,1,2,3,4,5,6,7,8,9,10,11,12,13,14,15);
}
// B-fragment (32x16 f16) from BT[N][K] rows: lane L holds col L&15; VGPR 0-3 K=(L>=16?8:0)+0..7, VGPR 4-7 K=+16.
__device__ __forceinline__ v16h ld_bfrag(const _Float16* p, int lane) {
    int kh = ((lane >> 4) << 3);                       // 0 or 8
    v8h lo = *(const v8h*)(p + kh);
    v8h hi = *(const v8h*)(p + 16 + kh);
    return __builtin_shufflevector(lo, hi, 0,1,2,3,4,5,6,7,8,9,10,11,12,13,14,15);
}

// ---------------- weight transpose + f32->f16 ----------------
__global__ __launch_bounds__(256) void wt_kernel(const float* __restrict__ w,
                                                 _Float16* __restrict__ wt,
                                                 int K, int N) {
    long i = (long)blockIdx.x * 256 + threadIdx.x;     // over K*N, w is [K][N]
    if (i < (long)K * N) {
        int k = (int)(i / N), n = (int)(i % N);
        wt[(size_t)n * K + k] = (_Float16)w[i];
    }
}

// ---------------- layernorm (row = 512) ----------------
__global__ __launch_bounds__(256) void ln_kernel(const float* __restrict__ x,
                                                 const float* __restrict__ g,
                                                 const float* __restrict__ b,
                                                 float* __restrict__ of32,      // nullable
                                                 _Float16* __restrict__ oh) {
    const int row = blockIdx.x;
    const int tid = threadIdx.x;
    const float* xr = x + (size_t)row * 512;
    float v0 = xr[tid], v1 = xr[tid + 256];
    float s  = v0 + v1;
    float s2 = v0 * v0 + v1 * v1;
    #pragma unroll
    for (int off = 16; off; off >>= 1) {
        s  += __shfl_down(s,  off, 32);
        s2 += __shfl_down(s2, off, 32);
    }
    __shared__ float sa[8], sb[8], bc[2];
    int lane = tid & 31, wid = tid >> 5;
    if (lane == 0) { sa[wid] = s; sb[wid] = s2; }
    __syncthreads();
    if (tid == 0) {
        float t = 0.f, t2 = 0.f;
        #pragma unroll
        for (int i = 0; i < 8; ++i) { t += sa[i]; t2 += sb[i]; }
        float m = t * (1.0f / 512.0f);
        float var = t2 * (1.0f / 512.0f) - m * m;
        bc[0] = m; bc[1] = rsqrtf(var + 1e-5f);
    }
    __syncthreads();
    float m = bc[0], rs = bc[1];
    float y0 = (v0 - m) * rs * g[tid]       + b[tid];
    float y1 = (v1 - m) * rs * g[tid + 256] + b[tid + 256];
    oh[(size_t)row * 512 + tid]       = (_Float16)y0;
    oh[(size_t)row * 512 + tid + 256] = (_Float16)y1;
    if (of32) {
        of32[(size_t)row * 512 + tid]       = y0;
        of32[(size_t)row * 512 + tid + 256] = y1;
    }
}

// ---------------- generic WMMA GEMM: C[M,N] = A[M,K] * BT[N,K]^T + bias ----------------
// BM=64, BN=128, BK=32; 256 threads = 8 waves (4 M x 2 N); each wave: 16 x 64.
// Double-buffered LDS, filled with async global->LDS B128 ops (ASYNCcnt).
// MODE 0: QKV scatter (q scaled)   MODE 1: proj + residual -> f32
// MODE 2: GELU(exact) -> f16       MODE 3: + residual -> f32 (res may alias outf)
#define A_STR 40
#define B_STR 40
template<int MODE>
__global__ __launch_bounds__(256) void gemm_kernel(const _Float16* __restrict__ A,
                                                   const _Float16* __restrict__ BT,
                                                   const float* __restrict__ bias,
                                                   int M, int N, int K,
                                                   float* __restrict__ outf,
                                                   _Float16* __restrict__ outh,
                                                   const float* __restrict__ res,
                                                   _Float16* __restrict__ qo,
                                                   _Float16* __restrict__ ko,
                                                   _Float16* __restrict__ vo) {
    __shared__ alignas(16) _Float16 Ash[2][64 * A_STR];
    __shared__ alignas(16) _Float16 Bsh[2][128 * B_STR];
    const int tid  = threadIdx.x;
    const int lane = tid & 31;
    const int wid  = tid >> 5;
    const int wm   = wid & 3;           // 0..3 -> M offset
    const int wn   = wid >> 2;          // 0..1 -> N offset
    const int m0   = blockIdx.x * 64;
    const int n0   = blockIdx.y * 128;

    // this thread's fixed chunk slots
    const int ar = tid >> 2, ac = tid & 3;                  // A: 256 chunks of 16B
    const int b0r = tid >> 2, b0c = tid & 3;                // B: 512 chunks
    const int b1r = (tid + 256) >> 2, b1c = tid & 3;

    v8f acc[4];
    #pragma unroll
    for (int t = 0; t < 4; ++t) acc[t] = (v8f){};

    // prologue fill buffer 0
    async_b128(&Ash[0][ar * A_STR + ac * 8],  A  + (size_t)(m0 + ar)  * K + ac * 8);
    async_b128(&Bsh[0][b0r * B_STR + b0c * 8], BT + (size_t)(n0 + b0r) * K + b0c * 8);
    async_b128(&Bsh[0][b1r * B_STR + b1c * 8], BT + (size_t)(n0 + b1r) * K + b1c * 8);
    wait_async0();
    __syncthreads();

    int buf = 0;
    for (int k0 = 0; k0 < K; k0 += 32) {
        int nk = k0 + 32;
        if (nk < K) {   // prefetch next tile into other buffer (async, overlaps WMMA)
            int nb = buf ^ 1;
            async_b128(&Ash[nb][ar * A_STR + ac * 8],  A  + (size_t)(m0 + ar)  * K + nk + ac * 8);
            async_b128(&Bsh[nb][b0r * B_STR + b0c * 8], BT + (size_t)(n0 + b0r) * K + nk + b0c * 8);
            async_b128(&Bsh[nb][b1r * B_STR + b1c * 8], BT + (size_t)(n0 + b1r) * K + nk + b1c * 8);
        }
        const _Float16* ap = &Ash[buf][(wm * 16 + (lane & 15)) * A_STR];
        v16h a = ld_afrag(ap, lane);
        #pragma unroll
        for (int t = 0; t < 4; ++t) {
            const _Float16* bp = &Bsh[buf][(wn * 64 + t * 16 + (lane & 15)) * B_STR];
            v16h bfr = ld_bfrag(bp, lane);
            acc[t] = WMMA_F16(a, bfr, acc[t]);
        }
        wait_async0();
        __syncthreads();
        buf ^= 1;
    }

    // epilogue: lane<16: N=lane, M=j; lane>=16: N=lane-16, M=j+8
    const int nl    = lane & 15;
    const int mbase = (lane >> 4) * 8;
    #pragma unroll
    for (int t = 0; t < 4; ++t) {
        int n = n0 + wn * 64 + t * 16 + nl;
        float bs = bias[n];
        #pragma unroll
        for (int j = 0; j < 8; ++j) {
            int m = m0 + wm * 16 + mbase + j;
            float val = acc[t][j] + bs;
            if (MODE == 0) {
                int which = n >> 9, hh = (n >> 6) & 7, d = n & 63;
                int bw = m >> 8, wq = m & 255;
                size_t idx = (((size_t)(bw * 8 + hh) * 256) + wq) * 64 + d;
                if      (which == 0) qo[idx] = (_Float16)(val * 0.125f);
                else if (which == 1) ko[idx] = (_Float16)val;
                else                 vo[idx] = (_Float16)val;
            } else if (MODE == 1) {
                size_t idx = (size_t)m * N + n;
                outf[idx] = val + res[idx];
            } else if (MODE == 2) {
                float gel = 0.5f * val * (1.0f + erff(val * 0.70710678f));
                outh[(size_t)m * N + n] = (_Float16)gel;
            } else {
                size_t idx = (size_t)m * N + n;
                outf[idx] = res[idx] + val;
            }
        }
    }
}

// ---------------- attention: one WG (128 thr = 4 waves) per (window, head) ----------------
#define Q_STR 72      // f16, row 144B (16B aligned), bank stride 36 dw -> conflict free
#define V_STR 280     // f16, row 560B
#define S_STR 260     // f32, row 1040B
#define P_STR 264     // f16, row 528B
__global__ __launch_bounds__(128) void attn_kernel(const _Float16* __restrict__ qh,
                                                   const _Float16* __restrict__ kh,
                                                   const _Float16* __restrict__ vh,
                                                   const float* __restrict__ rpb,
                                                   _Float16* __restrict__ outh) {
    __shared__ alignas(16) _Float16 Qs[256 * Q_STR];
    __shared__ alignas(16) _Float16 Ks[256 * Q_STR];
    __shared__ alignas(16) _Float16 Vt[64 * V_STR];
    __shared__ alignas(16) float    Sbuf[4 * 16 * S_STR];
    __shared__ alignas(16) _Float16 Pbuf[4 * 16 * P_STR];
    __shared__ float Rinv[4 * 16];

    const int bwh = blockIdx.x;           // bw*8 + h
    const int h   = bwh & 7;
    const int bw  = bwh >> 3;
    const int tid = threadIdx.x;

    // async fill of Q,K: 2048 chunks of 16B each (256 rows x 8 chunks)
    const _Float16* qg = qh + (size_t)bwh * 16384;
    const _Float16* kg = kh + (size_t)bwh * 16384;
    for (int i = tid; i < 2048; i += 128) {
        int r = i >> 3, cp = i & 7;
        async_b128(&Qs[r * Q_STR + cp * 8], qg + r * 64 + cp * 8);
        async_b128(&Ks[r * Q_STR + cp * 8], kg + r * 64 + cp * 8);
    }
    const _Float16* vg = vh + (size_t)bwh * 16384;
    for (int i = tid; i < 16384; i += 128) {         // transpose V -> Vt[d][key]
        int key = i >> 6, d = i & 63;
        Vt[d * V_STR + key] = vg[i];
    }
    wait_async0();
    __syncthreads();

    const int lane = tid & 31;
    const int wid  = tid >> 5;
    float*     Sw = Sbuf + wid * 16 * S_STR;
    _Float16*  Pw = Pbuf + wid * 16 * P_STR;
    float*     Rw = Rinv + wid * 16;
    const int nl    = lane & 15;
    const int mbase = (lane >> 4) * 8;

    for (int cc = 0; cc < 4; ++cc) {
        const int q0 = wid * 64 + cc * 16;
        const _Float16* qa = Qs + (q0 + nl) * Q_STR;
        v16h a0 = ld_afrag(qa, lane);
        v16h a1 = ld_afrag(qa + 32, lane);
        // S = q k^T (+bias)
        for (int nt = 0; nt < 16; ++nt) {
            const _Float16* kb = Ks + (nt * 16 + nl) * Q_STR;
            v16h b0 = ld_bfrag(kb, lane);
            v16h b1 = ld_bfrag(kb + 32, lane);
            v8f c = (v8f){};
            c = WMMA_F16(a0, b0, c);
            c = WMMA_F16(a1, b1, c);
            int n = nt * 16 + nl;                    // key index
            #pragma unroll
            for (int j = 0; j < 8; ++j) {
                int m = mbase + j;                   // row in chunk
                int q = q0 + m;                      // query index in window
                Sw[m * S_STR + n] = c[j] + rpb[(q - n + 255) * 8 + h];
            }
        }
        asm volatile("s_wait_dscnt 0" ::: "memory");
        // softmax: lanes 0..15 own one row each
        if (lane < 16) {
            float* Srow = Sw + lane * S_STR;
            float mx = -1e30f;
            for (int c2 = 0; c2 < 256; ++c2) mx = fmaxf(mx, Srow[c2]);
            float sum = 0.f;
            _Float16* Prow = Pw + lane * P_STR;
            for (int c2 = 0; c2 < 256; ++c2) {
                float e = __expf(Srow[c2] - mx);
                sum += e;
                Prow[c2] = (_Float16)e;
            }
            Rw[lane] = 1.0f / sum;
        }
        asm volatile("s_wait_dscnt 0" ::: "memory");
        // O = P V, scaled by 1/rowsum
        const _Float16* pa = Pw + nl * P_STR;
        for (int nt = 0; nt < 4; ++nt) {
            const _Float16* vb = Vt + (nt * 16 + nl) * V_STR;
            v8f o = (v8f){};
            #pragma unroll
            for (int kt = 0; kt < 8; ++kt)
                o = WMMA_F16(ld_afrag(pa + kt * 32, lane), ld_bfrag(vb + kt * 32, lane), o);
            int n = nt * 16 + nl;                    // dim within head
            #pragma unroll
            for (int j = 0; j < 8; ++j) {
                int m = mbase + j;
                float r = Rw[m];
                outh[(size_t)(bw * 256 + q0 + m) * 512 + h * 64 + n] = (_Float16)(o[j] * r);
            }
        }
        asm volatile("s_wait_dscnt 0" ::: "memory");
    }
}

// ---------------- launch ----------------
extern "C" void kernel_launch(void* const* d_in, const int* in_sizes, int n_in,
                              void* d_out, int out_size, void* d_ws, size_t ws_size,
                              hipStream_t stream) {
    (void)in_sizes; (void)n_in; (void)out_size; (void)ws_size;
    const float* x      = (const float*)d_in[0];
    const float* n1g    = (const float*)d_in[1];
    const float* n1b    = (const float*)d_in[2];
    const float* qkv_w  = (const float*)d_in[3];
    const float* qkv_b  = (const float*)d_in[4];
    const float* rpb    = (const float*)d_in[5];
    const float* proj_w = (const float*)d_in[6];
    const float* proj_b = (const float*)d_in[7];
    const float* n2g    = (const float*)d_in[8];
    const float* n2b    = (const float*)d_in[9];
    const float* fc1_w  = (const float*)d_in[10];
    const float* fc1_b  = (const float*)d_in[11];
    const float* fc2_w  = (const float*)d_in[12];
    const float* fc2_b  = (const float*)d_in[13];
    float* out = (float*)d_out;
    char*  ws  = (char*)d_ws;

    const int M = 32768;                 // B*L tokens
    _Float16* h1h   = (_Float16*)(ws + 0);                  // 32MB
    _Float16* q16   = (_Float16*)(ws + (size_t)33554432);
    _Float16* k16   = (_Float16*)(ws + (size_t)67108864);
    _Float16* v16   = (_Float16*)(ws + (size_t)100663296);
    _Float16* atth  = (_Float16*)(ws + (size_t)134217728);
    float*    x1f   = (float*)   (ws + (size_t)167772160);  // 64MB
    _Float16* h2h   = (_Float16*)(ws + (size_t)234881024);
    _Float16* mlph  = (_Float16*)(ws + (size_t)268435456);  // 128MB
    _Float16* wqkvT = (_Float16*)(ws + (size_t)402653184);
    _Float16* wprjT = (_Float16*)(ws + (size_t)404226048);
    _Float16* wfc1T = (_Float16*)(ws + (size_t)404750336);
    _Float16* wfc2T = (_Float16*)(ws + (size_t)406847488);

    wt_kernel<<<dim3((512 * 1536) / 256), 256, 0, stream>>>(qkv_w, wqkvT, 512, 1536);
    wt_kernel<<<dim3((512 * 512)  / 256), 256, 0, stream>>>(proj_w, wprjT, 512, 512);
    wt_kernel<<<dim3((512 * 2048) / 256), 256, 0, stream>>>(fc1_w,  wfc1T, 512, 2048);
    wt_kernel<<<dim3((2048 * 512) / 256), 256, 0, stream>>>(fc2_w,  wfc2T, 2048, 512);

    ln_kernel<<<dim3(M), 256, 0, stream>>>(x, n1g, n1b, nullptr, h1h);

    gemm_kernel<0><<<dim3(M / 64, 1536 / 128), 256, 0, stream>>>(
        h1h, wqkvT, qkv_b, M, 1536, 512, nullptr, nullptr, nullptr, q16, k16, v16);

    attn_kernel<<<dim3(1024), 128, 0, stream>>>(q16, k16, v16, rpb, atth);

    gemm_kernel<1><<<dim3(M / 64, 512 / 128), 256, 0, stream>>>(
        atth, wprjT, proj_b, M, 512, 512, x1f, nullptr, x, nullptr, nullptr, nullptr);

    ln_kernel<<<dim3(M), 256, 0, stream>>>(x1f, n2g, n2b, out, h2h);

    gemm_kernel<2><<<dim3(M / 64, 2048 / 128), 256, 0, stream>>>(
        h2h, wfc1T, fc1_b, M, 2048, 512, nullptr, mlph, nullptr, nullptr, nullptr, nullptr);

    gemm_kernel<3><<<dim3(M / 64, 512 / 128), 256, 0, stream>>>(
        mlph, wfc2T, fc2_b, M, 512, 2048, out, nullptr, out, nullptr, nullptr, nullptr);
}
